// AttentionBlock_30227979829446
// MI455X (gfx1250) — matrix operand order repaired and checked by reading, not compiled, not run on hardware
//
#include <hip/hip_runtime.h>

typedef __bf16 bf16_t;
typedef __attribute__((ext_vector_type(16))) __bf16 v16bf;
typedef __attribute__((ext_vector_type(8)))  __bf16 v8bf;
typedef __attribute__((ext_vector_type(8)))  float  v8f;

static constexpr int CC  = 256;   // channels
static constexpr int NN  = 4096;  // pixels (64*64)
static constexpr int BB  = 8;     // batch
static constexpr int GG  = 32;    // groups
static constexpr int CPG = CC / GG;

// ---------------- WMMA helpers (contiguous k order, same in A and B) ----------

union V16U { v16bf v; v8bf h[2]; };

// A-matrix 16x32 bf16 fragment. base points at row 0 of a [16 x row_stride]
// row-major region; lane L: row = L&15, lanes<16 hold K = k0+{0..15},
// lanes>=16 hold K = k0+{16..31}.
__device__ inline v16bf load_afrag(const bf16_t* base, int lane, int row_stride, int k0) {
  int r = lane & 15, hi = lane >> 4;
  const bf16_t* p = base + (size_t)r * row_stride + k0 + 16 * hi;
  V16U u;
  u.h[0] = *(const v8bf*)p;         // K = k0 + 16*hi + [0..7]
  u.h[1] = *(const v8bf*)(p + 8);   // K = k0 + 16*hi + 8 + [0..7]
  return u.v;
}

// B-matrix 32x16 bf16 fragment. base points at "column 0" where each B-column
// is a contiguous-K row of the array (stride col_stride); lane L: column =
// L&15, K = k0 + 16*(L>>4) + [0..15] contiguous.
__device__ inline v16bf load_bfrag(const bf16_t* base, int lane, int col_stride, int k0) {
  int cc = lane & 15, hi = lane >> 4;
  const bf16_t* p = base + (size_t)cc * col_stride + k0 + 16 * hi;
  return *(const v16bf*)p;
}

__device__ inline v8f wmma_bf16(v16bf a, v16bf b, v8f c) {
  return __builtin_amdgcn_wmma_f32_16x16x32_bf16(false, a, false, b, (short)0, c, false, false);
}

// ---------------- GroupNorm ------------------------------------------------

__global__ void gn_stats_kernel(const float* __restrict__ x, float* __restrict__ stats) {
  int bg = blockIdx.x;              // b*GG + g
  int b = bg / GG, g = bg % GG;
  const float* base = x + ((size_t)b * CC + (size_t)g * CPG) * NN;
  float s = 0.f, ss = 0.f;
  for (int i = threadIdx.x; i < CPG * NN; i += 256) {
    float v = base[i];
    s += v; ss += v * v;
  }
  __shared__ float sh[512];
  sh[threadIdx.x] = s; sh[256 + threadIdx.x] = ss;
  __syncthreads();
  for (int o = 128; o > 0; o >>= 1) {
    if ((int)threadIdx.x < o) {
      sh[threadIdx.x]       += sh[threadIdx.x + o];
      sh[256 + threadIdx.x] += sh[256 + threadIdx.x + o];
    }
    __syncthreads();
  }
  if (threadIdx.x == 0) {
    const float inv = 1.0f / (float)(CPG * NN);
    float mean = sh[0] * inv;
    float var  = sh[256] * inv - mean * mean;
    stats[bg * 2 + 0] = mean;
    stats[bg * 2 + 1] = rsqrtf(var + 1e-5f);
  }
}

// normalize + scale/bias, write transposed bf16 [b][n][c] (c contiguous)
__global__ void gn_apply_kernel(const float* __restrict__ x,
                                const float* __restrict__ gsc,
                                const float* __restrict__ gbi,
                                const float* __restrict__ stats,
                                bf16_t* __restrict__ xnT) {
  size_t idx = (size_t)blockIdx.x * 256 + threadIdx.x;  // [b][n][c] linear
  int c = (int)(idx & (CC - 1));
  size_t bn = idx >> 8;
  int n = (int)(bn & (NN - 1));
  int b = (int)(bn >> 12);
  const float* st = stats + (b * GG + c / CPG) * 2;
  float mean = st[0], rstd = st[1];
  float v = x[((size_t)b * CC + c) * NN + n];
  v = (v - mean) * rstd * gsc[c] + gbi[c];
  xnT[idx] = (bf16_t)v;
}

__global__ void cvt_bf16_kernel(const float* __restrict__ src, bf16_t* __restrict__ dst, int n) {
  int i = blockIdx.x * 256 + threadIdx.x;
  if (i < n) dst[i] = (bf16_t)src[i];
}

// ---------------- fused Q/K/V projection (WMMA GEMM) -----------------------
// D[o][n] = sum_c W[o][c] * xnT[n][c]; one wave per 16x16 output tile.
// q,k written bf16 [b][n][o]; v written bf16 [b][o][n]; q prescaled by 1/16.
__global__ __launch_bounds__(32) void qkv_proj_kernel(
    const bf16_t* __restrict__ xnT,
    const bf16_t* __restrict__ wqb, const bf16_t* __restrict__ wkb, const bf16_t* __restrict__ wvb,
    const float* __restrict__ bq, const float* __restrict__ bk, const float* __restrict__ bv,
    bf16_t* __restrict__ qT, bf16_t* __restrict__ kT, bf16_t* __restrict__ vO) {
  int lane = threadIdx.x;
  int b = blockIdx.z, o0 = blockIdx.y * 16, n0 = blockIdx.x * 16;
  const bf16_t* xb = xnT + ((size_t)b * NN + n0) * CC;   // 16 pixel-rows, c contiguous
  v8f accq = {}, acck = {}, accv = {};
#pragma unroll
  for (int k0 = 0; k0 < CC; k0 += 32) {
    v16bf bf = load_bfrag(xb, lane, CC, k0);
    v16bf aq = load_afrag(wqb + (size_t)o0 * CC, lane, CC, k0);
    v16bf ak = load_afrag(wkb + (size_t)o0 * CC, lane, CC, k0);
    v16bf av = load_afrag(wvb + (size_t)o0 * CC, lane, CC, k0);
    accq = wmma_bf16(aq, bf, accq);
    acck = wmma_bf16(ak, bf, acck);
    accv = wmma_bf16(av, bf, accv);
  }
  int nn = lane & 15, hi = lane >> 4;
  int n = n0 + nn;
  bf16_t* qrow = qT + ((size_t)b * NN + n) * CC + o0 + 8 * hi;
  bf16_t* krow = kT + ((size_t)b * NN + n) * CC + o0 + 8 * hi;
#pragma unroll
  for (int j = 0; j < 8; ++j) {
    int o = o0 + 8 * hi + j;
    qrow[j] = (bf16_t)((accq[j] + bq[o]) * 0.0625f);   // fold 1/sqrt(C)
    krow[j] = (bf16_t)(acck[j] + bk[o]);
    vO[((size_t)b * CC + o) * NN + n] = (bf16_t)(accv[j] + bv[o]);
  }
}

// ---------------- flash attention ------------------------------------------
// One wave per (batch, 16-row block). Online softmax over 4096 keys in
// 32-column steps; accumulator = 16 rows x 256 channels fp32 (16 v8f tiles).
__global__ __launch_bounds__(32) void flash_attn_kernel(
    const bf16_t* __restrict__ qT, const bf16_t* __restrict__ kT,
    const bf16_t* __restrict__ vI, bf16_t* __restrict__ attnT) {
  int lane = threadIdx.x;
  int b = blockIdx.y;
  int n0 = blockIdx.x * 16;
  __shared__ bf16_t pbuf[16 * 32];   // P tile staging (D-layout -> A-layout)

  const bf16_t* qbase = qT + ((size_t)b * NN + n0) * CC;
  v16bf qf[8];
#pragma unroll
  for (int kk = 0; kk < 8; ++kk) qf[kk] = load_afrag(qbase, lane, CC, kk * 32);

  v8f acc[16];
#pragma unroll
  for (int t = 0; t < 16; ++t) acc[t] = (v8f){};
  float mrow[8], lrow[8];
#pragma unroll
  for (int j = 0; j < 8; ++j) { mrow[j] = -1e30f; lrow[j] = 0.f; }

  const bf16_t* kbase = kT + (size_t)b * NN * CC;
  const bf16_t* vbase = vI + (size_t)b * CC * NN;
  int mm = lane & 15, hi = lane >> 4;

  for (int m0 = 0; m0 < NN; m0 += 32) {
    if (m0 + 32 < NN) __builtin_prefetch(kbase + (size_t)(m0 + 32) * CC, 0, 1);
    // ---- scores: two 16-col tiles, K = 256 in 8 WMMA steps each
    v8f s0 = {}, s1 = {};
#pragma unroll
    for (int kk = 0; kk < 8; ++kk) {
      v16bf k0f = load_bfrag(kbase + (size_t)m0 * CC,        lane, CC, kk * 32);
      v16bf k1f = load_bfrag(kbase + (size_t)(m0 + 16) * CC, lane, CC, kk * 32);
      s0 = wmma_bf16(qf[kk], k0f, s0);
      s1 = wmma_bf16(qf[kk], k1f, s1);
    }
    // ---- online softmax (row stats across 16 lanes of each half-wave)
    float scl[8];
#pragma unroll
    for (int j = 0; j < 8; ++j) {
      float t = fmaxf(s0[j], s1[j]);
      t = fmaxf(t, __shfl_xor(t, 8, 16));
      t = fmaxf(t, __shfl_xor(t, 4, 16));
      t = fmaxf(t, __shfl_xor(t, 2, 16));
      t = fmaxf(t, __shfl_xor(t, 1, 16));
      float mnew = fmaxf(mrow[j], t);
      scl[j] = __expf(mrow[j] - mnew);
      float p0 = __expf(s0[j] - mnew);
      float p1 = __expf(s1[j] - mnew);
      s0[j] = p0; s1[j] = p1;
      float r = p0 + p1;
      r += __shfl_xor(r, 8, 16);
      r += __shfl_xor(r, 4, 16);
      r += __shfl_xor(r, 2, 16);
      r += __shfl_xor(r, 1, 16);
      lrow[j] = lrow[j] * scl[j] + r;
      mrow[j] = mnew;
    }
    // ---- P tile: D-layout regs -> LDS row-major [16][32]
    __syncthreads();
#pragma unroll
    for (int j = 0; j < 8; ++j) {
      pbuf[(j + 8 * hi) * 32 + mm]      = (bf16_t)s0[j];
      pbuf[(j + 8 * hi) * 32 + 16 + mm] = (bf16_t)s1[j];
    }
    __syncthreads();
    v16bf pf = load_afrag(pbuf, lane, 32, 0);
    // ---- rescale accumulator and add P (16x32) x V (32x16) per c-tile
#pragma unroll
    for (int t = 0; t < 16; ++t) {
#pragma unroll
      for (int j = 0; j < 8; ++j) acc[t][j] *= scl[j];
      v16bf vf = load_bfrag(vbase + (size_t)t * 16 * NN, lane, NN, m0);
      acc[t] = wmma_bf16(pf, vf, acc[t]);
    }
  }
  // ---- epilogue: divide by l, store bf16 [b][n][c]
  float rln[8];
#pragma unroll
  for (int j = 0; j < 8; ++j) rln[j] = 1.0f / lrow[j];
#pragma unroll
  for (int t = 0; t < 16; ++t) {
    bf16_t* dst = attnT + ((size_t)b * NN + n0) * CC + t * 16 + mm;
#pragma unroll
    for (int j = 0; j < 8; ++j)
      dst[(j + 8 * hi) * CC] = (bf16_t)(acc[t][j] * rln[j]);
  }
}

// ---------------- output projection + residual -----------------------------
__global__ __launch_bounds__(32) void out_proj_kernel(
    const bf16_t* __restrict__ attnT, const bf16_t* __restrict__ wob,
    const float* __restrict__ bo, const float* __restrict__ x,
    float* __restrict__ out) {
  int lane = threadIdx.x;
  int b = blockIdx.z, o0 = blockIdx.y * 16, n0 = blockIdx.x * 16;
  const bf16_t* bbase = attnT + ((size_t)b * NN + n0) * CC;
  v8f acc = {};
#pragma unroll
  for (int k0 = 0; k0 < CC; k0 += 32) {
    v16bf af = load_afrag(wob + (size_t)o0 * CC, lane, CC, k0);
    v16bf bf = load_bfrag(bbase, lane, CC, k0);
    acc = wmma_bf16(af, bf, acc);
  }
  int nn = lane & 15, hi = lane >> 4;
#pragma unroll
  for (int j = 0; j < 8; ++j) {
    int o = o0 + 8 * hi + j;
    size_t idx = ((size_t)b * CC + o) * NN + n0 + nn;
    out[idx] = x[idx] + acc[j] + bo[o];
  }
}

// ---------------- launch ---------------------------------------------------

extern "C" void kernel_launch(void* const* d_in, const int* in_sizes, int n_in,
                              void* d_out, int out_size, void* d_ws, size_t ws_size,
                              hipStream_t stream) {
  const float* x   = (const float*)d_in[0];
  const float* gsc = (const float*)d_in[1];
  const float* gbi = (const float*)d_in[2];
  const float* wq  = (const float*)d_in[3];
  const float* bq  = (const float*)d_in[4];
  const float* wk  = (const float*)d_in[5];
  const float* bk  = (const float*)d_in[6];
  const float* wv  = (const float*)d_in[7];
  const float* bv  = (const float*)d_in[8];
  const float* wo  = (const float*)d_in[9];
  const float* bo  = (const float*)d_in[10];
  float* out = (float*)d_out;

  const size_t ACT = (size_t)BB * NN * CC * sizeof(bf16_t);  // 16 MB
  char* ws = (char*)d_ws;
  bf16_t* xnT = (bf16_t*)(ws + 0 * ACT);
  bf16_t* qT  = (bf16_t*)(ws + 1 * ACT);
  bf16_t* kT  = (bf16_t*)(ws + 2 * ACT);
  bf16_t* vv  = (bf16_t*)(ws + 3 * ACT);
  bf16_t* aT  = (bf16_t*)(ws + 4 * ACT);
  bf16_t* wqb = (bf16_t*)(ws + 5 * ACT);
  bf16_t* wkb = wqb + CC * CC;
  bf16_t* wvb = wkb + CC * CC;
  bf16_t* wob = wvb + CC * CC;
  float*  stats = (float*)(wob + CC * CC);

  gn_stats_kernel<<<BB * GG, 256, 0, stream>>>(x, stats);
  gn_apply_kernel<<<(BB * (size_t)NN * CC) / 256, 256, 0, stream>>>(x, gsc, gbi, stats, xnT);
  cvt_bf16_kernel<<<CC * CC / 256, 256, 0, stream>>>(wq, wqb, CC * CC);
  cvt_bf16_kernel<<<CC * CC / 256, 256, 0, stream>>>(wk, wkb, CC * CC);
  cvt_bf16_kernel<<<CC * CC / 256, 256, 0, stream>>>(wv, wvb, CC * CC);
  cvt_bf16_kernel<<<CC * CC / 256, 256, 0, stream>>>(wo, wob, CC * CC);

  qkv_proj_kernel<<<dim3(NN / 16, CC / 16, BB), 32, 0, stream>>>(
      xnT, wqb, wkb, wvb, bq, bk, bv, qT, kT, vv);
  flash_attn_kernel<<<dim3(NN / 16, BB), 32, 0, stream>>>(qT, kT, vv, aT);
  out_proj_kernel<<<dim3(NN / 16, CC / 16, BB), 32, 0, stream>>>(aT, wob, bo, x, out);
}
